// AnyVariateAttention_33285996544529
// MI455X (gfx1250) — matrix operand order repaired and checked by reading, not compiled, not run on hardware
//
#include <hip/hip_runtime.h>
#include <hip/hip_bf16.h>

// ---------------------------------------------------------------------------
// AnyVariateAttention for MI455X (gfx1250): bf16 WMMA GEMMs + flash attention.
// ---------------------------------------------------------------------------

#define D_MODEL 512
#define NUM_HEADS 8
#define HEAD_DIM 64
#define BATCH 4
#define SEQ 2048
#define M_ROWS (BATCH * SEQ)   // 8192

typedef __attribute__((ext_vector_type(16))) __bf16 v16bf;
typedef __attribute__((ext_vector_type(8)))  float  v8f;

union BF16Frag { v16bf v; uint4 q[2]; unsigned short u[16]; };
union F8       { v8f   v; float f[8]; };

__device__ __forceinline__ unsigned short bf16_bits(float f) {
    union { __bf16 h; unsigned short u; } cv;
    cv.h = (__bf16)f;
    return cv.u;
}

__device__ __forceinline__ v8f wmma_bf16(v16bf a, v16bf b, v8f c) {
    return __builtin_amdgcn_wmma_f32_16x16x32_bf16(
        /*neg_a=*/false, a, /*neg_b=*/false, b,
        /*c_mod=*/(short)0, c, /*reuse_a=*/false, /*reuse_b=*/false);
}

// A fragment (16x32 bf16, row-major source, leading dim = ld elements).
// Lane l: row m = l&15; half = l>>4 selects K base (0 or 16);
// VGPR0-7 hold K = kb..kb+15 (16 consecutive K values).
__device__ __forceinline__ v16bf load_frag_a(const __bf16* __restrict__ base, int ld) {
    const int l  = threadIdx.x & 31;
    const int m  = l & 15;
    const int kb = (l >> 4) << 4;
    const __bf16* p = base + (size_t)m * ld + kb;
    BF16Frag f;
    f.q[0] = *(const uint4*)(p);
    f.q[1] = *(const uint4*)(p + 8);
    return f.v;
}

// B fragment (32x16 bf16) loaded from B^T in row-major (rows = N, contiguous K).
// Lane l: col n = l&15; K base = (l>>4)*16; 16 consecutive K values.
__device__ __forceinline__ v16bf load_frag_b(const __bf16* __restrict__ baseT, int ld) {
    const int l  = threadIdx.x & 31;
    const int n  = l & 15;
    const int kb = (l >> 4) << 4;
    const __bf16* p = baseT + (size_t)n * ld + kb;
    BF16Frag f;
    f.q[0] = *(const uint4*)(p);
    f.q[1] = *(const uint4*)(p + 8);
    return f.v;
}

// 16 consecutive bf16 from LDS (for B fragments staged in shared memory).
__device__ __forceinline__ v16bf frag16(const unsigned short* p) {
    BF16Frag f;
    f.q[0] = *(const uint4*)(p);
    f.q[1] = *(const uint4*)(p + 8);
    return f.v;
}

// ---------------------------------------------------------------------------
// fp32 -> bf16 elementwise convert
// ---------------------------------------------------------------------------
__global__ void f32_to_bf16_kernel(const float* __restrict__ src,
                                   __bf16* __restrict__ dst, int n) {
    int i = blockIdx.x * blockDim.x + threadIdx.x;
    if (i < n) dst[i] = (__bf16)src[i];
}

// ---------------------------------------------------------------------------
// GEMM: out = X(bf16, MxK row-major) * W^T(W is NxK row-major, bf16) + bias.
// Block = 8 waves; block owns one 16-row M tile (staged in LDS once), each
// wave computes a 16x64 strip (4 WMMA accumulators reusing one A fragment).
// mode 0: Q proj + RoPE -> [B,H,S,Hd] bf16
// mode 1: K proj + RoPE -> [B,H,S,Hd] bf16
// mode 2: V proj        -> [B,H,Hd,S] bf16 (transposed for P*V B-fragments)
// mode 3: O proj        -> [M, D] fp32 (final output)
// ---------------------------------------------------------------------------
#define A_STRIDE 520   // 512 + 8 pad: fragment lanes hit distinct LDS banks

__global__ void gemm_wmma_kernel(const __bf16* __restrict__ X,
                                 const __bf16* __restrict__ W,
                                 const float*  __restrict__ bias,
                                 __bf16* __restrict__ outB,
                                 float*  __restrict__ outF,
                                 int mode) {
    __shared__ __align__(16) unsigned short Alds[16][A_STRIDE];

    const int tid   = threadIdx.x;
    const int wave  = tid >> 5;
    const int l     = tid & 31;
    const int half  = l >> 4;
    const int tileM = blockIdx.x << 4;     // 16 rows of X per block
    const int tileN = wave << 6;           // 64 output cols per wave

    // ---- stage the 16x512 A tile into LDS (64B per thread) -----------------
    {
        const int r = tid >> 4;            // 16 rows
        const int c = (tid & 15) << 5;     // 16 segments of 32 shorts (64B)
        const uint4* src = (const uint4*)(X + (size_t)(tileM + r) * D_MODEL + c);
        uint4* dst = (uint4*)&Alds[r][c];
#pragma unroll
        for (int i = 0; i < 4; ++i) dst[i] = src[i];
    }
    __syncthreads();

    F8 acc[4];
#pragma unroll
    for (int t = 0; t < 4; ++t) acc[t].v = (v8f){};

    const int am = l & 15;
    const int akb = (l >> 4) << 4;

#pragma unroll 2
    for (int k = 0; k < D_MODEL; k += 32) {
        // A fragment from LDS
        BF16Frag fa;
        const unsigned short* ar = &Alds[am][k + akb];
        fa.q[0] = *(const uint4*)(ar);
        fa.q[1] = *(const uint4*)(ar + 8);
        const v16bf a = fa.v;
#pragma unroll
        for (int t = 0; t < 4; ++t) {
            v16bf b = load_frag_b(W + (size_t)(tileN + t * 16) * D_MODEL + k, D_MODEL);
            acc[t].v = wmma_bf16(a, b, acc[t].v);
        }
    }

    // ---- epilogue -----------------------------------------------------------
#pragma unroll
    for (int t = 0; t < 4; ++t) {
        const int ncol = tileN + t * 16 + (l & 15);
        const float bv = bias[ncol];
        float vals[8];
#pragma unroll
        for (int e = 0; e < 8; ++e) vals[e] = acc[t].f[e] + bv;

        if (mode <= 1) {
            // RoPE: column pair (2j,2j+1) uses angle index a = (2j) mod 32 ->
            // a = hd & 30; the partner column lives in lane l^1.
            const int hd   = ncol & 63;
            const int aidx = hd & 30;
            const float inv_freq = __powf(10000.0f, -(float)aidx / 32.0f);
            const bool even = (hd & 1) == 0;
#pragma unroll
            for (int e = 0; e < 8; ++e) {
                const int mg = tileM + e + (half << 3);
                const int s  = mg & (SEQ - 1);
                float sn, cs;
                __sincosf((float)s * inv_freq, &sn, &cs);
                float partner = __shfl_xor(vals[e], 1, 32);
                vals[e] = even ? (vals[e] * cs - partner * sn)
                               : (partner * sn + vals[e] * cs);
            }
        }

        if (mode == 3) {
#pragma unroll
            for (int e = 0; e < 8; ++e) {
                const int mg = tileM + e + (half << 3);
                outF[(size_t)mg * D_MODEL + ncol] = vals[e];
            }
        } else {
            const int h  = ncol >> 6;
            const int hd = ncol & 63;
#pragma unroll
            for (int e = 0; e < 8; ++e) {
                const int mg = tileM + e + (half << 3);
                const int b  = mg >> 11;          // / SEQ
                const int s  = mg & (SEQ - 1);
                if (mode == 2) {
                    // V transposed: [B,H,Hd,S]
                    outB[(((size_t)(b * NUM_HEADS + h) * HEAD_DIM + hd) * SEQ) + s] =
                        (__bf16)vals[e];
                } else {
                    // Q/K: [B,H,S,Hd]
                    outB[(((size_t)(b * NUM_HEADS + h) * SEQ + s) * HEAD_DIM) + hd] =
                        (__bf16)vals[e];
                }
            }
        }
    }
}

// ---------------------------------------------------------------------------
// Flash attention: grid (SEQ/128, B*H), block 256 (8 waves).
// Each wave owns 16 query rows. K/V tiles for each 32-key block are staged
// once per block into double-buffered LDS and shared by all 8 waves.
// ---------------------------------------------------------------------------
#define KS_STRIDE 72   // 64 + 8 pad (bank step 36 mod 64 -> distinct banks)
#define VS_STRIDE 40   // 32 + 8 pad (bank step 20 mod 64 -> distinct banks)

__global__ void flash_attn_kernel(const __bf16* __restrict__ Q,
                                  const __bf16* __restrict__ K,
                                  const __bf16* __restrict__ Vt,
                                  const long long* __restrict__ vid,
                                  const float* __restrict__ mask,
                                  const float* __restrict__ u_same,
                                  const float* __restrict__ u_cross,
                                  __bf16* __restrict__ Obuf) {
    __shared__ __align__(16) unsigned short Ksh[2][32][KS_STRIDE];  // 32 keys x 64 dims
    __shared__ __align__(16) unsigned short Vsh[2][64][VS_STRIDE];  // 64 dims x 32 keys
    __shared__ __align__(16) unsigned short Plds[8][16][32];        // per-wave P tile

    const int tid  = threadIdx.x;
    const int wave = tid >> 5;
    const int l    = tid & 31;
    const int half = l >> 4;
    const int bh   = blockIdx.y;
    const int b    = bh >> 3;
    const int h    = bh & 7;
    const int q0   = blockIdx.x * 128 + wave * 16;

    const float scale = 0.125f;      // 1/sqrt(64)
    const float us = u_same[h];
    const float uc = u_cross[h];

    const __bf16* Qb = Q  + (size_t)bh * SEQ * HEAD_DIM;
    const __bf16* Kb = K  + (size_t)bh * SEQ * HEAD_DIM;
    const __bf16* Vb = Vt + (size_t)bh * HEAD_DIM * SEQ;
    const long long* vidb = vid  + (size_t)b * SEQ;
    const float*    maskb = mask + (size_t)b * SEQ;

    // Q tile fragments: 16 queries x 64 dims = 2 A fragments (K=0..31, 32..63)
    const v16bf aq0 = load_frag_a(Qb + (size_t)q0 * HEAD_DIM + 0,  HEAD_DIM);
    const v16bf aq1 = load_frag_a(Qb + (size_t)q0 * HEAD_DIM + 32, HEAD_DIM);

    long long vq[8];
#pragma unroll
    for (int e = 0; e < 8; ++e) vq[e] = vidb[q0 + e + (half << 3)];

    float mrun[8], lrun[8];
    F8 acc[4];
#pragma unroll
    for (int e = 0; e < 8; ++e) { mrun[e] = -1e30f; lrun[e] = 0.0f; }
#pragma unroll
    for (int t = 0; t < 4; ++t) acc[t].v = (v8f){};

    // cooperative K/V tile stage: 256 threads move 4KB + 4KB per 32-key block
    auto stage = [&](int bufn, int kbn) {
        {
            const int r = tid >> 3, seg = (tid & 7) << 3;          // 32 rows x 8 segs
            *(uint4*)&Ksh[bufn][r][seg] =
                *(const uint4*)(Kb + (size_t)(kbn + r) * HEAD_DIM + seg);
        }
        {
            const int r = tid >> 2, seg = (tid & 3) << 3;          // 64 rows x 4 segs
            *(uint4*)&Vsh[bufn][r][seg] =
                *(const uint4*)(Vb + (size_t)r * SEQ + kbn + seg);
        }
    };

    stage(0, 0);
    int buf = 0;

    for (int kb = 0; kb < SEQ; kb += 32) {
        __syncthreads();                         // staged tiles visible
        if (kb + 32 < SEQ) stage(buf ^ 1, kb + 32);
        if (kb + 64 < SEQ)
            __builtin_prefetch(Kb + (size_t)(kb + 64) * HEAD_DIM, 0, 1);

        // ---- S = Q * K^T for two 16-key sub-tiles (from LDS) ---------------
        const int kd   = (l >> 4) << 4;          // dim sub-block within fragment
        const int keyr = l & 15;
        F8 s0, s1; s0.v = (v8f){}; s1.v = (v8f){};
        s0.v = wmma_bf16(aq0, frag16(&Ksh[buf][keyr][kd]),           s0.v);
        s0.v = wmma_bf16(aq1, frag16(&Ksh[buf][keyr][32 + kd]),      s0.v);
        s1.v = wmma_bf16(aq0, frag16(&Ksh[buf][16 + keyr][kd]),      s1.v);
        s1.v = wmma_bf16(aq1, frag16(&Ksh[buf][16 + keyr][32 + kd]), s1.v);

        // ---- scale + variate bias + mask, online softmax -------------------
        const int k0c = kb + (l & 15);
        const int k1c = k0c + 16;
        const long long vk0 = vidb[k0c];
        const long long vk1 = vidb[k1c];
        const float mt0 = (1.0f - maskb[k0c]) * -1.0e9f;
        const float mt1 = (1.0f - maskb[k1c]) * -1.0e9f;

        float p0[8], p1[8], alpha[8];
#pragma unroll
        for (int e = 0; e < 8; ++e) {
            float x0 = s0.f[e] * scale + ((vq[e] == vk0) ? us : uc) + mt0;
            float x1 = s1.f[e] * scale + ((vq[e] == vk1) ? us : uc) + mt1;
            float rm = fmaxf(x0, x1);
#pragma unroll
            for (int off = 1; off < 16; off <<= 1)
                rm = fmaxf(rm, __shfl_xor(rm, off, 32));
            const float mnew = fmaxf(mrun[e], rm);
            const float al = __expf(mrun[e] - mnew);
            p0[e] = __expf(x0 - mnew);
            p1[e] = __expf(x1 - mnew);
            float rs = p0[e] + p1[e];
#pragma unroll
            for (int off = 1; off < 16; off <<= 1)
                rs += __shfl_xor(rs, off, 32);
            lrun[e] = lrun[e] * al + rs;
            mrun[e] = mnew;
            alpha[e] = al;
        }
#pragma unroll
        for (int t = 0; t < 4; ++t)
#pragma unroll
            for (int e = 0; e < 8; ++e) acc[t].f[e] *= alpha[e];

        // ---- P (C layout) -> per-wave LDS -> A fragment --------------------
        {
            const int col = l & 15;
#pragma unroll
            for (int e = 0; e < 8; ++e) {
                const int row = e + (half << 3);
                Plds[wave][row][col]      = bf16_bits(p0[e]);
                Plds[wave][row][col + 16] = bf16_bits(p1[e]);
            }
        }
        asm volatile("s_wait_dscnt 0" ::: "memory");

        v16bf ap;
        {
            const unsigned short* prow = &Plds[wave][l & 15][0];
            const int kb2 = (l >> 4) << 4;
            BF16Frag f;
            f.q[0] = *(const uint4*)(prow + kb2);
            f.q[1] = *(const uint4*)(prow + kb2 + 8);
            ap = f.v;
        }

        // ---- O += P * V (V tile in LDS, rows = hd, contiguous keys) --------
        const int vkb = (l >> 4) << 4;
#pragma unroll
        for (int t = 0; t < 4; ++t) {
            v16bf bv = frag16(&Vsh[buf][t * 16 + (l & 15)][vkb]);
            acc[t].v = wmma_bf16(ap, bv, acc[t].v);
        }

        buf ^= 1;
    }

    // ---- normalize + store to [B,S,D] bf16 for the output projection ------
    float inv[8];
#pragma unroll
    for (int e = 0; e < 8; ++e) inv[e] = 1.0f / lrun[e];

#pragma unroll
    for (int t = 0; t < 4; ++t) {
        const int col = h * HEAD_DIM + t * 16 + (l & 15);
#pragma unroll
        for (int e = 0; e < 8; ++e) {
            const int s = q0 + e + (half << 3);
            Obuf[((size_t)b * SEQ + s) * D_MODEL + col] =
                (__bf16)(acc[t].f[e] * inv[e]);
        }
    }
}

// ---------------------------------------------------------------------------
// Host-side orchestration
// ---------------------------------------------------------------------------
extern "C" void kernel_launch(void* const* d_in, const int* in_sizes, int n_in,
                              void* d_out, int out_size, void* d_ws, size_t ws_size,
                              hipStream_t stream) {
    const float*     x    = (const float*)d_in[0];
    const long long* vid  = (const long long*)d_in[1];   // int64 variate ids
    const float*     mask = (const float*)d_in[2];
    const float*     Wq   = (const float*)d_in[3];
    const float*     bq   = (const float*)d_in[4];
    const float*     Wk   = (const float*)d_in[5];
    const float*     bk   = (const float*)d_in[6];
    const float*     Wv   = (const float*)d_in[7];
    const float*     bv   = (const float*)d_in[8];
    const float*     Wo   = (const float*)d_in[9];
    const float*     bo   = (const float*)d_in[10];
    const float*     us   = (const float*)d_in[11];
    const float*     ucx  = (const float*)d_in[12];
    float* out = (float*)d_out;

    // Workspace layout (bf16 buffers)
    char* ws = (char*)d_ws;
    const size_t xN = (size_t)M_ROWS * D_MODEL;          // 8192*512
    const size_t wN = (size_t)D_MODEL * D_MODEL;         // 512*512
    __bf16* xb   = (__bf16*)ws;                 ws += xN * 2;
    __bf16* Wqb  = (__bf16*)ws;                 ws += wN * 2;
    __bf16* Wkb  = (__bf16*)ws;                 ws += wN * 2;
    __bf16* Wvb  = (__bf16*)ws;                 ws += wN * 2;
    __bf16* Wob  = (__bf16*)ws;                 ws += wN * 2;
    __bf16* Qb   = (__bf16*)ws;                 ws += xN * 2;
    __bf16* Kb   = (__bf16*)ws;                 ws += xN * 2;
    __bf16* Vtb  = (__bf16*)ws;                 ws += xN * 2;
    __bf16* Obuf = (__bf16*)ws;                 ws += xN * 2;

    // 1) fp32 -> bf16 converts
    {
        int n = (int)xN;
        f32_to_bf16_kernel<<<(n + 255) / 256, 256, 0, stream>>>(x, xb, n);
        n = (int)wN;
        f32_to_bf16_kernel<<<(n + 255) / 256, 256, 0, stream>>>(Wq, Wqb, n);
        f32_to_bf16_kernel<<<(n + 255) / 256, 256, 0, stream>>>(Wk, Wkb, n);
        f32_to_bf16_kernel<<<(n + 255) / 256, 256, 0, stream>>>(Wv, Wvb, n);
        f32_to_bf16_kernel<<<(n + 255) / 256, 256, 0, stream>>>(Wo, Wob, n);
    }

    // 2) QKV projections (+RoPE for Q,K; transpose store for V)
    dim3 ggrid(M_ROWS / 16, 1);
    gemm_wmma_kernel<<<ggrid, 256, 0, stream>>>(xb, Wqb, bq, Qb,  nullptr, 0);
    gemm_wmma_kernel<<<ggrid, 256, 0, stream>>>(xb, Wkb, bk, Kb,  nullptr, 1);
    gemm_wmma_kernel<<<ggrid, 256, 0, stream>>>(xb, Wvb, bv, Vtb, nullptr, 2);

    // 3) flash attention
    dim3 fgrid(SEQ / 128, BATCH * NUM_HEADS);
    flash_attn_kernel<<<fgrid, 256, 0, stream>>>(Qb, Kb, Vtb, vid, mask, us, ucx, Obuf);

    // 4) output projection -> fp32 result
    gemm_wmma_kernel<<<ggrid, 256, 0, stream>>>(Obuf, Wob, bo, nullptr, out, 3);
}
